// ContrastiveLoss_18399639896831
// MI455X (gfx1250) — matrix operand order repaired and checked by reading, not compiled, not run on hardware
//
#include <hip/hip_runtime.h>
#include <hip/hip_bf16.h>

// ---- CDNA5 WMMA vector types --------------------------------------------
typedef __attribute__((ext_vector_type(16))) __bf16 v16bf;
typedef __attribute__((ext_vector_type(8)))  __bf16 v8bf;
typedef __attribute__((ext_vector_type(8)))  float  v8f;

#define BM 128
#define BN 256
#define BK 32
#define KSTEPS 16          // D / BK with D = 512
#define LDSTR 40           // 32 + 8 bf16 pad per row (80 B) -> conflict-free frag reads
#define ABUF (BM * LDSTR)  // one A buffer, bf16 elems
#define BBUF (BN * LDSTR)  // one B buffer, bf16 elems

// pack two f32 -> two bf16 (round-to-nearest-even) in one dword
__device__ __forceinline__ unsigned bf16pk(float a, float b) {
    unsigned ua = __float_as_uint(a); ua += 0x7FFFu + ((ua >> 16) & 1u);
    unsigned ub = __float_as_uint(b); ub += 0x7FFFu + ((ub >> 16) & 1u);
    return (ua >> 16) | (ub & 0xFFFF0000u);
}

__global__ void cl_init_kernel(float* accum) {
    if (threadIdx.x < 4) accum[threadIdx.x] = 0.0f;
}

// sim = x1 * x2^T on bf16 WMMA, fused contrastive-loss masking + reduction.
// Block: 256 threads = 8 wave32s (2 x 4), wave tile 64x64, block tile 128x256.
// Double-buffered LDS with register prefetch: 1 barrier per K-step.
__global__ __launch_bounds__(256) void cl_gemm_loss_kernel(
    const float* __restrict__ x1, const int* __restrict__ lab1,
    const float* __restrict__ x2, const int* __restrict__ lab2,
    float* __restrict__ accum, int D)
{
    __shared__ __attribute__((aligned(16))) __bf16 sA[2 * ABUF];
    __shared__ __attribute__((aligned(16))) __bf16 sB[2 * BBUF];
    __shared__ int    sL1[BM];
    __shared__ int    sL2[BN];
    __shared__ float4 sRed[256];

    const int t    = threadIdx.x;
    const int lane = t & 31;
    const int wid  = t >> 5;        // 8 waves
    const int h    = lane >> 4;     // half-wave select
    const int l15  = lane & 15;
    const int wm   = wid >> 2;      // 0..1 -> 64 rows each
    const int wn   = wid & 3;       // 0..3 -> 64 cols each
    const int blockM = blockIdx.y * BM;
    const int blockN = blockIdx.x * BN;

    // stage the label slices this block needs
    if (t < BM) sL1[t] = lab1[blockM + t];
    sL2[t] = lab2[blockN + t];

    // per-thread staging geometry: 8 float4 per row, rows strided by 32 per chunk
    const int r0 = t >> 3;          // 0..31
    const int c4 = t & 7;           // float4 column
    const float* gA = x1 + (blockM + r0) * D + c4 * 4;   // + p*32*D + k0
    const float* gB = x2 + (blockN + r0) * D + c4 * 4;

    v8f acc[4][4];
#pragma unroll
    for (int i = 0; i < 4; ++i)
#pragma unroll
        for (int j = 0; j < 4; ++j)
            acc[i][j] = (v8f){0.f,0.f,0.f,0.f,0.f,0.f,0.f,0.f};

    float4 ra[4];   // A prefetch registers (128x32 tile, 4 slots/thread)
    float4 rb[8];   // B prefetch registers (256x32 tile, 8 slots/thread)

    // prefetch K-step 0
#pragma unroll
    for (int p = 0; p < 4; ++p) ra[p] = *(const float4*)(gA + p * 32 * D);
#pragma unroll
    for (int p = 0; p < 8; ++p) rb[p] = *(const float4*)(gB + p * 32 * D);

    for (int ks = 0; ks < KSTEPS; ++ks) {
        __bf16* bufA = sA + (ks & 1) * ABUF;
        __bf16* bufB = sB + (ks & 1) * BBUF;

        // convert staged registers -> bf16 LDS tile
#pragma unroll
        for (int p = 0; p < 4; ++p) {
            uint2 pk; pk.x = bf16pk(ra[p].x, ra[p].y); pk.y = bf16pk(ra[p].z, ra[p].w);
            *(uint2*)(&bufA[(r0 + p * 32) * LDSTR + c4 * 4]) = pk;
        }
#pragma unroll
        for (int p = 0; p < 8; ++p) {
            uint2 pk; pk.x = bf16pk(rb[p].x, rb[p].y); pk.y = bf16pk(rb[p].z, rb[p].w);
            *(uint2*)(&bufB[(r0 + p * 32) * LDSTR + c4 * 4]) = pk;
        }
        __syncthreads();   // single barrier per K-step (see ordering argument)

        // issue next tile's global loads; they land during the WMMA phase
        if (ks + 1 < KSTEPS) {
            const int k0 = (ks + 1) * BK;
#pragma unroll
            for (int p = 0; p < 4; ++p) ra[p] = *(const float4*)(gA + p * 32 * D + k0);
#pragma unroll
            for (int p = 0; p < 8; ++p) rb[p] = *(const float4*)(gB + p * 32 * D + k0);
        }

        // ---- B fragments: 16-bit B 32x16 layout ----
        // lane (h,l15): column n = l15, elem j<8 -> K = h*8 + j ; elem j>=8 -> K = 16 + h*8 + (j-8)
        v16bf bfrag[4];
#pragma unroll
        for (int j = 0; j < 4; ++j) {
            int rowb = wn * 64 + j * 16 + l15;   // row of x2 == column of B
            v8bf lo = *(const v8bf*)(&bufB[rowb * LDSTR + h * 8]);
            v8bf hi = *(const v8bf*)(&bufB[rowb * LDSTR + 16 + h * 8]);
#pragma unroll
            for (int e = 0; e < 8; ++e) { bfrag[j][e] = lo[e]; bfrag[j][8 + e] = hi[e]; }
        }
        // ---- A fragments (16-bit A 16x32 layout) + 16 WMMAs ----
        // lane (h,l15): elem j<8 -> K = h*8 + j ; elem j>=8 -> K = 16 + h*8 + (j-8)
#pragma unroll
        for (int i = 0; i < 4; ++i) {
            int row = wm * 64 + i * 16 + l15;
            v8bf lo = *(const v8bf*)(&bufA[row * LDSTR + h * 8]);
            v8bf hi = *(const v8bf*)(&bufA[row * LDSTR + 16 + h * 8]);
            v16bf afrag;
#pragma unroll
            for (int e = 0; e < 8; ++e) { afrag[e] = lo[e]; afrag[8 + e] = hi[e]; }
#pragma unroll
            for (int j = 0; j < 4; ++j)
                acc[i][j] = __builtin_amdgcn_wmma_f32_16x16x32_bf16(
                    false, afrag, false, bfrag[j], (short)0, acc[i][j], false, false);
        }
    }

    // ---- fused masking epilogue ----
    // C/D layout: lane l, elem r -> M = r + 8*(l>>4), N = l&15 (within 16x16 tile)
    const float POS_THR = 1.0f - 1e-6f - 0.05f;
    float pl = 0.f, nl = 0.f, pc = 0.f, nc = 0.f;
#pragma unroll
    for (int i = 0; i < 4; ++i) {
#pragma unroll
        for (int j = 0; j < 4; ++j) {
            int ncol = wn * 64 + j * 16 + l15;
            int l2v  = sL2[ncol];
#pragma unroll
            for (int r = 0; r < 8; ++r) {
                int mrow = wm * 64 + i * 16 + r + 8 * h;
                int l1v  = sL1[mrow];
                float s  = acc[i][j][r];
                bool valid = l1v > 0;
                bool same  = (l1v == l2v);
                if (valid &&  same && s < POS_THR) { pl += 1.0f - s; pc += 1.0f; }
                if (valid && !same && s > 0.5f)    { nl += s;        nc += 1.0f; }
            }
        }
    }

    __syncthreads();   // all frag reads done before sRed aliases nothing (separate array, but keep order clean)
    sRed[t] = make_float4(pl, nl, pc, nc);
    __syncthreads();
#pragma unroll
    for (int off = 128; off > 0; off >>= 1) {
        if (t < off) {
            float4 a = sRed[t], b = sRed[t + off];
            sRed[t] = make_float4(a.x + b.x, a.y + b.y, a.z + b.z, a.w + b.w);
        }
        __syncthreads();
    }
    if (t == 0) {
        float4 v = sRed[0];
        atomicAdd(accum + 0, v.x);
        atomicAdd(accum + 1, v.y);
        atomicAdd(accum + 2, v.z);
        atomicAdd(accum + 3, v.w);
    }
}

__global__ void cl_finalize_kernel(const int* __restrict__ lab1, int N,
                                   const float* __restrict__ accum,
                                   float* __restrict__ out)
{
    __shared__ float s[256];
    int t = threadIdx.x;
    float cnt = 0.f;
    for (int i = t; i < N; i += 256) cnt += (lab1[i] > 0) ? 1.0f : 0.0f;
    s[t] = cnt;
    __syncthreads();
    for (int off = 128; off > 0; off >>= 1) {
        if (t < off) s[t] += s[t + off];
        __syncthreads();
    }
    if (t == 0) {
        float n  = s[0];
        float pl = accum[0], nl = accum[1], pc = accum[2], nc = accum[3];
        out[0] = (pl + nl) / n;
        out[1] = nc / n;
        out[2] = rintf(100.0f * pc / n) / 100.0f;   // jnp.round = half-to-even
    }
}

extern "C" void kernel_launch(void* const* d_in, const int* in_sizes, int n_in,
                              void* d_out, int out_size, void* d_ws, size_t ws_size,
                              hipStream_t stream) {
    const float* x1  = (const float*)d_in[0];
    const int*  lab1 = (const int*)d_in[1];
    const float* x2  = (const float*)d_in[2];
    const int*  lab2 = (const int*)d_in[3];
    float* out   = (float*)d_out;
    float* accum = (float*)d_ws;     // 4 floats: pos_loss, neg_loss, pos_cnt, neg_cnt

    const int N = in_sizes[1];           // 8192
    const int M = in_sizes[3];           // 8192
    const int D = in_sizes[0] / N;       // 512

    cl_init_kernel<<<1, 64, 0, stream>>>(accum);

    dim3 grid(M / BN, N / BM);           // 32 x 64 blocks
    cl_gemm_loss_kernel<<<grid, 256, 0, stream>>>(x1, lab1, x2, lab2, accum, D);

    cl_finalize_kernel<<<1, 256, 0, stream>>>(lab1, N, accum, out);
}
